// Qwen3Attention_7748121002093
// MI455X (gfx1250) — matrix operand order repaired and checked by reading, not compiled, not run on hardware
//
#include <hip/hip_runtime.h>
#include <hip/hip_bf16.h>
#include <math.h>

typedef __bf16 bf16_t;
typedef __attribute__((ext_vector_type(16))) __bf16 v16bf;
typedef __attribute__((ext_vector_type(8)))  __bf16 v8bf;
typedef __attribute__((ext_vector_type(8)))  float  v8f;

#define B_    2
#define S_    2048
#define HID_  2048
#define H_    16
#define HKV_  4
#define D_    128
#define M_    (B_ * S_)                 // 4096
#define OQKV_ ((H_ + 2 * HKV_) * D_)    // 3072
#define NHEAD_TOT_ (H_ + 2 * HKV_)      // 24

// ---------------------------------------------------------------------------
// WMMA helper: D = A(16x32 bf16) * B(32x16 bf16) + C(16x16 f32)
// ---------------------------------------------------------------------------
__device__ __forceinline__ v8f wmma_bf16(v16bf a, v16bf b, v8f c) {
    return __builtin_amdgcn_wmma_f32_16x16x32_bf16(
        /*neg_a=*/false, a, /*neg_b=*/false, b,
        /*c_mod=*/(short)0, c, /*reuse_a=*/false, /*reuse_b=*/false);
}

// A-fragment (16 rows x 32 K) from row-major [rows, ld] bf16.
// Layout used here (contiguous k, not ISA table order): lanes 0-15 hold row M=lane, K=0..15;
//             lanes 16-31 hold row M=lane-16, K=16..31.
__device__ __forceinline__ v16bf load_a_frag(const bf16_t* base, int ld,
                                             int m0, int k0, int lane) {
    int half = lane >> 4;
    int row  = m0 + (lane & 15);
    const bf16_t* p = base + (size_t)row * ld + k0 + half * 16;
    v8bf lo = *(const v8bf*)(p);
    v8bf hi = *(const v8bf*)(p + 8);
    v16bf out;
#pragma unroll
    for (int i = 0; i < 8; ++i) { out[i] = lo[i]; out[i + 8] = hi[i]; }
    return out;
}

// B-fragment (32 K x 16 cols) from K-major source: row-major [cols, ld] bf16
// (i.e. element (k,n) at base[n*ld + k]).
// Layout used (contiguous k, matches A): lane n<16 holds col N=n, K=e+0..15; lane n+16 holds K=16..31.
__device__ __forceinline__ v16bf load_b_frag(const bf16_t* base, int ld,
                                             int n0, int k0, int lane) {
    int half = lane >> 4;
    int col  = n0 + (lane & 15);
    const bf16_t* p = base + (size_t)col * ld + k0 + half * 16;
    return *(const v16bf*)(p);   // 32 contiguous bytes
}

// ---------------------------------------------------------------------------
// fp32 -> bf16 conversion (grid-stride)
// ---------------------------------------------------------------------------
__global__ void cvt_bf16_kernel(const float* __restrict__ in,
                                bf16_t* __restrict__ out, int n) {
    int i = blockIdx.x * blockDim.x + threadIdx.x;
    int stride = gridDim.x * blockDim.x;
    for (; i < n; i += stride) out[i] = (bf16_t)in[i];
}

// ---------------------------------------------------------------------------
// GEMM: C[M,N] (f32) = A[M,K] (bf16, row-major) * B[N,K]^T (bf16, K-major)
// block = 256 threads = 8 waves arranged 4(M) x 2(N); wave tile = 32 x 64:
//   8 WMMA per 32-wide K-step against 12 b128 loads (2 A-frags + 4 B-frags).
// ~140 VGPRs/wave -> no spills with waves_per_eu = 1.
// grid = (M/128, N/128)
// ---------------------------------------------------------------------------
__global__ void __launch_bounds__(256, 1)
gemm_bf16_kernel(const bf16_t* __restrict__ A,
                 const bf16_t* __restrict__ Bw,
                 float* __restrict__ C,
                 int Mdim, int Ndim, int Kdim) {
    int lane = threadIdx.x & 31;
    int wave = threadIdx.x >> 5;
    int mw = wave & 3, nw = wave >> 2;           // 4 x 2 wave grid
    int m0 = blockIdx.x * 128 + mw * 32;
    int n0 = blockIdx.y * 128 + nw * 64;

    v8f acc[2][4] = {};                          // [mi][ni] 16x16 tiles
    for (int k0 = 0; k0 < Kdim; k0 += 32) {
        // prefetch one K-tile ahead (speculative; OOB silently dropped)
        __builtin_prefetch(A + (size_t)(m0 + (lane & 15)) * Kdim + k0 + 32, 0, 1);
        __builtin_prefetch(Bw + (size_t)(n0 + (lane & 15)) * Kdim + k0 + 32, 0, 1);

        v16bf a0 = load_a_frag(A, Kdim, m0,      k0, lane);
        v16bf a1 = load_a_frag(A, Kdim, m0 + 16, k0, lane);
#pragma unroll
        for (int ni = 0; ni < 4; ++ni) {
            v16bf b = load_b_frag(Bw, Kdim, n0 + ni * 16, k0, lane);
            acc[0][ni] = wmma_bf16(a0, b, acc[0][ni]);
            acc[1][ni] = wmma_bf16(a1, b, acc[1][ni]);
        }
    }

    int half = lane >> 4, nl = lane & 15;
#pragma unroll
    for (int mi = 0; mi < 2; ++mi)
#pragma unroll
        for (int ni = 0; ni < 4; ++ni)
#pragma unroll
            for (int i = 0; i < 8; ++i) {
                int m = m0 + mi * 16 + i + 8 * half;
                int n = n0 + ni * 16 + nl;
                C[(size_t)m * Ndim + n] = acc[mi][ni][i];
            }
}

// ---------------------------------------------------------------------------
// Fused per-head LayerNorm + RoPE; splits qkv into q/k (bf16, [b,h,s,d]) and
// v (bf16, transposed [b,hk,d,s] so P*V B-fragments are contiguous).
// One wave per (b, s, head); lane handles 4 consecutive d.
// ---------------------------------------------------------------------------
__global__ void lnrope_kernel(const float* __restrict__ qkv,
                              const float* __restrict__ qg,
                              const float* __restrict__ qb,
                              const float* __restrict__ kg,
                              const float* __restrict__ kbias,
                              bf16_t* __restrict__ qh,
                              bf16_t* __restrict__ kh,
                              bf16_t* __restrict__ vt) {
    int lane = threadIdx.x & 31;
    int wave = threadIdx.x >> 5;
    int w = blockIdx.x * 8 + wave;
    int head = w % NHEAD_TOT_;
    int bs   = w / NHEAD_TOT_;
    int b = bs / S_, s = bs % S_;

    const float* row = qkv + (size_t)bs * OQKV_ + head * D_;
    int d0 = lane * 4;
    float4 x4 = *(const float4*)(row + d0);
    float v[4] = {x4.x, x4.y, x4.z, x4.w};

    if (head < H_ + HKV_) {
        // ---- LayerNorm over D=128 (full-wave reduction) ----
        float sum = v[0] + v[1] + v[2] + v[3];
#pragma unroll
        for (int o = 1; o < 32; o <<= 1) sum += __shfl_xor(sum, o, 32);
        float mu = sum * (1.0f / 128.0f);
        float vs = 0.f;
#pragma unroll
        for (int j = 0; j < 4; ++j) { float d = v[j] - mu; vs += d * d; }
#pragma unroll
        for (int o = 1; o < 32; o <<= 1) vs += __shfl_xor(vs, o, 32);
        float rstd = rsqrtf(vs * (1.0f / 128.0f) + 1e-5f);

        const float* g  = (head < H_) ? qg : kg;
        const float* bb = (head < H_) ? qb : kbias;
        float y[4];
#pragma unroll
        for (int j = 0; j < 4; ++j)
            y[j] = (v[j] - mu) * rstd * g[d0 + j] + bb[d0 + j];

        // ---- RoPE: partner element lives 64 elems away = lane^16 ----
#pragma unroll
        for (int j = 0; j < 4; ++j) {
            float partner = __shfl_xor(y[j], 16, 32);
            float rot = (lane < 16) ? -partner : partner;
            int   fi  = (d0 + j) & 63;
            // inv_freq = base^(-fi/64);  ln(1e6) = 13.8155105579643
            float inv = __expf(-(float)fi * (13.815510557964274f / 64.0f));
            float ang = (float)s * inv;
            float sn, cs;
            __sincosf(ang, &sn, &cs);
            v[j] = y[j] * cs + rot * sn;
        }

        if (head < H_) {
            bf16_t* dst = qh + (((size_t)(b * H_ + head) * S_ + s) * D_ + d0);
#pragma unroll
            for (int j = 0; j < 4; ++j) dst[j] = (bf16_t)v[j];
        } else {
            int hk = head - H_;
            bf16_t* dst = kh + (((size_t)(b * HKV_ + hk) * S_ + s) * D_ + d0);
#pragma unroll
            for (int j = 0; j < 4; ++j) dst[j] = (bf16_t)v[j];
        }
    } else {
        // ---- V head: just convert, store d-major ([b,hk,d,s]) ----
        int hv = head - H_ - HKV_;
        bf16_t* base = vt + (size_t)(b * HKV_ + hv) * D_ * S_;
#pragma unroll
        for (int j = 0; j < 4; ++j)
            base[(size_t)(d0 + j) * S_ + s] = (bf16_t)v[j];
    }
}

// ---------------------------------------------------------------------------
// Causal flash attention. One wave per 16 query rows of one (b,h).
// grid = (S/16, B*H), block = 32.
// scores via WMMA (Q 16xD * K^T), online softmax, P*V via WMMA after an LDS
// C->A layout transpose of the probability tile.
// ---------------------------------------------------------------------------
__global__ void __launch_bounds__(32, 1)
attn_kernel(const bf16_t* __restrict__ qh,
            const bf16_t* __restrict__ kh,
            const bf16_t* __restrict__ vt,
            bf16_t* __restrict__ attn) {
    int lane = threadIdx.x;
    int qblk = blockIdx.x;
    int bh   = blockIdx.y;
    int b = bh / H_, h = bh % H_;
    int hk = h / (H_ / HKV_);          // GQA: repeat -> kv head = h / 4
    int q0 = qblk * 16;

    const bf16_t* qbase = qh + (size_t)(b * H_ + h) * S_ * D_;
    const bf16_t* kbase = kh + (size_t)(b * HKV_ + hk) * S_ * D_;
    const bf16_t* vbase = vt + (size_t)(b * HKV_ + hk) * D_ * S_;

    int half = lane >> 4, nl = lane & 15;
    const float scale = 0.08838834764831845f;   // 1/sqrt(128)

    // Q fragments for the whole D=128 (kept in registers)
    v16bf qf[4];
#pragma unroll
    for (int t = 0; t < 4; ++t) qf[t] = load_a_frag(qbase, D_, q0, t * 32, lane);

    v8f o[8] = {};                      // 16 x 128 output accumulator
    float mrow[8], lrow[8];
#pragma unroll
    for (int i = 0; i < 8; ++i) { mrow[i] = -1e30f; lrow[i] = 0.f; }

    __shared__ __align__(16) bf16_t pP[16 * 32];

    for (int kb = 0; kb < q0 + 16; kb += 32) {
        // ---- scores: two 16x16 tiles covering 32 keys ----
        v8f sc[2] = {};
#pragma unroll
        for (int tn = 0; tn < 2; ++tn)
#pragma unroll
            for (int t = 0; t < 4; ++t) {
                v16bf kf = load_b_frag(kbase, D_, kb + tn * 16, t * 32, lane);
                sc[tn] = wmma_bf16(qf[t], kf, sc[tn]);
            }

        // ---- mask, online softmax, rescale O ----
        float pm0[8], pm1[8];
#pragma unroll
        for (int i = 0; i < 8; ++i) {
            int m = q0 + i + 8 * half;
            float x0 = sc[0][i] * scale;
            float x1 = sc[1][i] * scale;
            if (kb + nl      > m) x0 = -1e30f;
            if (kb + 16 + nl > m) x1 = -1e30f;
            float mx = fmaxf(x0, x1);
#pragma unroll
            for (int off = 1; off < 16; off <<= 1)
                mx = fmaxf(mx, __shfl_xor(mx, off, 32));
            float nm = fmaxf(mrow[i], mx);
            float e0 = __expf(x0 - nm);
            float e1 = __expf(x1 - nm);
            pm0[i] = e0; pm1[i] = e1;
            float rs = e0 + e1;
#pragma unroll
            for (int off = 1; off < 16; off <<= 1)
                rs += __shfl_xor(rs, off, 32);
            float alpha = __expf(mrow[i] - nm);
            lrow[i] = lrow[i] * alpha + rs;
            mrow[i] = nm;
#pragma unroll
            for (int t = 0; t < 8; ++t) o[t][i] *= alpha;
        }

        // ---- C-layout -> A-layout transpose of P through LDS ----
#pragma unroll
        for (int i = 0; i < 8; ++i) {
            int m = i + 8 * half;
            pP[m * 32 + nl]      = (bf16_t)pm0[i];
            pP[m * 32 + 16 + nl] = (bf16_t)pm1[i];
        }
        __syncthreads();   // single-wave WG: S_NOP, but keeps DS ordering explicit
        v16bf pf;
        {
            const bf16_t* p = pP + (size_t)nl * 32 + half * 16;
            v8bf lo = *(const v8bf*)(p);
            v8bf hi = *(const v8bf*)(p + 8);
#pragma unroll
            for (int i = 0; i < 8; ++i) { pf[i] = lo[i]; pf[i + 8] = hi[i]; }
        }
        __syncthreads();

        // ---- O += P(16x32) * V(32x128): V is d-major so B-frags contiguous ----
#pragma unroll
        for (int t = 0; t < 8; ++t) {
            const bf16_t* vp = vbase + (size_t)(t * 16 + nl) * S_ + kb + half * 16;
            v16bf vf = *(const v16bf*)vp;
            o[t] = wmma_bf16(pf, vf, o[t]);
        }
    }

    // ---- epilogue: normalize and store [B,S,H*D] bf16 for the output GEMM ----
#pragma unroll
    for (int t = 0; t < 8; ++t)
#pragma unroll
        for (int i = 0; i < 8; ++i) {
            int m = q0 + i + 8 * half;
            int d = t * 16 + nl;
            float val = o[t][i] / lrow[i];
            attn[(((size_t)b * S_ + m) * H_ + h) * D_ + d] = (bf16_t)val;
        }
}

// ---------------------------------------------------------------------------
// Host-side orchestration
// ---------------------------------------------------------------------------
extern "C" void kernel_launch(void* const* d_in, const int* in_sizes, int n_in,
                              void* d_out, int out_size, void* d_ws, size_t ws_size,
                              hipStream_t stream) {
    const float* x     = (const float*)d_in[0];   // [B,S,HID]
    const float* w_qkv = (const float*)d_in[1];   // [OQKV, HID]
    const float* w_o   = (const float*)d_in[2];   // [HID, H*D]
    const float* qn_g  = (const float*)d_in[3];
    const float* qn_b  = (const float*)d_in[4];
    const float* kn_g  = (const float*)d_in[5];
    const float* kn_b  = (const float*)d_in[6];
    float* out = (float*)d_out;                   // [B,S,HID]

    // workspace carve-up (all 256B aligned by construction)
    char* ws = (char*)d_ws;
    size_t off = 0;
    bf16_t* xb    = (bf16_t*)(ws + off); off += (size_t)M_ * HID_ * 2;       // 16 MB
    bf16_t* wqkvb = (bf16_t*)(ws + off); off += (size_t)OQKV_ * HID_ * 2;    // 12 MB
    bf16_t* wob   = (bf16_t*)(ws + off); off += (size_t)HID_ * HID_ * 2;     //  8 MB
    float*  qkv   = (float*) (ws + off); off += (size_t)M_ * OQKV_ * 4;      // 48 MB
    bf16_t* qhp   = (bf16_t*)(ws + off); off += (size_t)B_ * H_   * S_ * D_ * 2; // 16 MB
    bf16_t* khp   = (bf16_t*)(ws + off); off += (size_t)B_ * HKV_ * S_ * D_ * 2; //  4 MB
    bf16_t* vtp   = (bf16_t*)(ws + off); off += (size_t)B_ * HKV_ * S_ * D_ * 2; //  4 MB
    bf16_t* attnb = (bf16_t*)(ws + off); off += (size_t)M_ * HID_ * 2;       // 16 MB

    // 1) fp32 -> bf16 conversions
    cvt_bf16_kernel<<<2048, 256, 0, stream>>>(x,     xb,    M_ * HID_);
    cvt_bf16_kernel<<<2048, 256, 0, stream>>>(w_qkv, wqkvb, OQKV_ * HID_);
    cvt_bf16_kernel<<<2048, 256, 0, stream>>>(w_o,   wob,   HID_ * HID_);

    // 2) QKV projection: [4096,3072] = xb[4096,2048] * wqkvb[3072,2048]^T
    gemm_bf16_kernel<<<dim3(M_ / 128, OQKV_ / 128), 256, 0, stream>>>(
        xb, wqkvb, qkv, M_, OQKV_, HID_);

    // 3) fused LayerNorm + RoPE + split/transpose
    lnrope_kernel<<<(M_ * NHEAD_TOT_) / 8, 256, 0, stream>>>(
        qkv, qn_g, qn_b, kn_g, kn_b, qhp, khp, vtp);

    // 4) causal flash attention
    attn_kernel<<<dim3(S_ / 16, B_ * H_), 32, 0, stream>>>(qhp, khp, vtp, attnb);

    // 5) output projection: [4096,2048] = attnb[4096,2048] * wob[2048,2048]^T
    gemm_bf16_kernel<<<dim3(M_ / 128, HID_ / 128), 256, 0, stream>>>(
        attnb, wob, out, M_, HID_, HID_);
}
